// LSTMDecoder_41798621724870
// MI455X (gfx1250) — compile-verified, matrix operand order checked by reading, not run on hardware
//
#include <hip/hip_runtime.h>
#include <hip/hip_bf16.h>

// Problem constants (match reference)
#define B_   128
#define DIN  512
#define H_   1024
#define O_   1024
#define T_   256
#define G4H  (4 * H_)

typedef __attribute__((ext_vector_type(16))) __bf16 bf16x16;
typedef __attribute__((ext_vector_type(8)))  float  floatx8;

__device__ __forceinline__ unsigned short f2bf(float f) {
    union { float f; unsigned u; } v; v.f = f;
    unsigned r = v.u + 0x7FFFu + ((v.u >> 16) & 1u);   // round-to-nearest-even
    return (unsigned short)(r >> 16);
}
__device__ __forceinline__ float sigm(float x)  { return 1.0f / (1.0f + __expf(-x)); }
__device__ __forceinline__ float tanhfa(float x){ return 1.0f - 2.0f / (__expf(2.0f * x) + 1.0f); }

__device__ __forceinline__ bf16x16 ldbf(const unsigned short* p) {
    return *reinterpret_cast<const bf16x16*>(p);
}

// ---------------------------------------------------------------------------
// h0 = z@W_fc_h.T + b_fc_h  (-> bf16, becomes step-0 A matrix)
// c0 = z@W_fc_c.T + b_fc_c  (-> fp32 working c buffer)
// ---------------------------------------------------------------------------
__global__ void init_state_kernel(const float* __restrict__ z,
                                  const float* __restrict__ Wfh, const float* __restrict__ bfh,
                                  const float* __restrict__ Wfc, const float* __restrict__ bfc,
                                  unsigned short* __restrict__ h0_bf, float* __restrict__ c) {
    int id = blockIdx.x * blockDim.x + threadIdx.x;
    if (id >= B_ * H_) return;
    int b = id / H_, j = id % H_;
    const float* zr = z + b * DIN;
    const float* wh = Wfh + j * DIN;
    const float* wc = Wfc + j * DIN;
    float ah = bfh[j], ac = bfc[j];
#pragma unroll 4
    for (int k = 0; k < DIN; ++k) { float zv = zr[k]; ah += zv * wh[k]; ac += zv * wc[k]; }
    h0_bf[id] = f2bf(ah);
    c[id]     = ac;
}

// ---------------------------------------------------------------------------
// u[n]  = x0 . W_ih[n,:] + b_ih[n] + b_hh[n]   (step-0 gate bias; x0 is one row)
// bc[n] = b_ih[n] + b_hh[n]                    (steps >= 1 gate bias)
// ---------------------------------------------------------------------------
__global__ void init_gatebias_kernel(const float* __restrict__ x0, const float* __restrict__ Wih,
                                     const float* __restrict__ bih, const float* __restrict__ bhh,
                                     float* __restrict__ u, float* __restrict__ bc) {
    int n = blockIdx.x * blockDim.x + threadIdx.x;
    if (n >= G4H) return;
    float bb = bih[n] + bhh[n];
    const float* w = Wih + n * H_;
    float acc = bb;
#pragma unroll 4
    for (int k = 0; k < H_; ++k) acc += x0[k] * w[k];
    u[n] = acc; bc[n] = bb;
}

// Wc = bf16(W_ih + W_hh)  [steps>=1];  Wh = bf16(W_hh)  [step 0]
__global__ void conv_rec_weights_kernel(const float* __restrict__ Wih, const float* __restrict__ Whh,
                                        unsigned short* __restrict__ Wc, unsigned short* __restrict__ Wh) {
    int id = blockIdx.x * blockDim.x + threadIdx.x;
    if (id >= G4H * H_) return;
    float a = Wih[id], b = Whh[id];
    Wc[id] = f2bf(a + b);
    Wh[id] = f2bf(b);
}

__global__ void conv_out_weights_kernel(const float* __restrict__ Wout, unsigned short* __restrict__ Wo) {
    int id = blockIdx.x * blockDim.x + threadIdx.x;
    if (id >= O_ * H_) return;
    Wo[id] = f2bf(Wout[id]);
}

// ---------------------------------------------------------------------------
// One LSTM step: gates = A @ W.T + bias, then elementwise LSTM cell.
// Grid: 16 blocks (64 h-columns each), 512 threads = 16 waves.
// Wave (wm, wh): 2 M-tiles (32 batch rows) x 16 h-columns; it computes the
// i/f/g/o tiles for the SAME h-columns so the cell math stays in-register.
// ---------------------------------------------------------------------------
__global__ __launch_bounds__(512)
void lstm_step_kernel(const unsigned short* __restrict__ A,    // [B_,H_] bf16 (h_prev or h0)
                      const unsigned short* __restrict__ W,    // [4H,H] bf16 (Wc or Whh)
                      const float* __restrict__ bias,          // [4H]     (bc or u)
                      float* __restrict__ c,                   // [B_,H_] fp32 (in place)
                      unsigned short* __restrict__ hs,         // [T,B,H] bf16 history
                      unsigned short* __restrict__ hnext,      // [B_,H_] bf16 next h
                      int t) {
    const int w    = threadIdx.x >> 5;
    const int lane = threadIdx.x & 31;
    const int half = lane >> 4, lrow = lane & 15;
    const int m_base    = (w & 3) * 32;                       // 2 M-tiles
    const int hcol_base = blockIdx.x * 64 + (w >> 2) * 16;    // 16 h-columns

    floatx8 acc[4][2];
#pragma unroll
    for (int g = 0; g < 4; ++g) { acc[g][0] = floatx8{}; acc[g][1] = floatx8{}; }

    const int krow = 16 * half;
    const unsigned short* a0p = A + (m_base + lrow) * H_ + krow;
    const unsigned short* a1p = a0p + 16 * H_;
    const unsigned short* bp[4];
#pragma unroll
    for (int g = 0; g < 4; ++g) bp[g] = W + (g * H_ + hcol_base + lrow) * H_ + krow;

    for (int k0 = 0; k0 < H_; k0 += 32) {
        bf16x16 a0 = ldbf(a0p + k0);
        bf16x16 a1 = ldbf(a1p + k0);
        __builtin_prefetch(a0p + k0 + 128, 0, 1);
#pragma unroll
        for (int g = 0; g < 4; ++g) {
            bf16x16 bm = ldbf(bp[g] + k0);
            __builtin_prefetch(bp[g] + k0 + 128, 0, 1);
            acc[g][0] = __builtin_amdgcn_wmma_f32_16x16x32_bf16(false, a0, false, bm, (short)0, acc[g][0], false, false);
            acc[g][1] = __builtin_amdgcn_wmma_f32_16x16x32_bf16(false, a1, false, bm, (short)0, acc[g][1], false, false);
        }
    }

    const int hcol = hcol_base + lrow;                        // D-tile: n = lane%16
    const float bi = bias[0 * H_ + hcol];
    const float bf = bias[1 * H_ + hcol];
    const float bg = bias[2 * H_ + hcol];
    const float bo = bias[3 * H_ + hcol];
    unsigned short* hsrow = hs + (size_t)t * B_ * H_;

#pragma unroll
    for (int mt = 0; mt < 2; ++mt) {
#pragma unroll
        for (int r = 0; r < 8; ++r) {
            int m = m_base + 16 * mt + 8 * half + r;          // D-tile row mapping
            size_t idx = (size_t)m * H_ + hcol;
            float gi = acc[0][mt][r] + bi;
            float gf = acc[1][mt][r] + bf;
            float gg = acc[2][mt][r] + bg;
            float go = acc[3][mt][r] + bo;
            float cp = c[idx];
            float cn = sigm(gf) * cp + sigm(gi) * tanhfa(gg);
            float hn = sigm(go) * tanhfa(cn);
            c[idx] = cn;
            unsigned short hb = f2bf(hn);
            hsrow[idx] = hb;
            hnext[idx] = hb;
        }
    }
}

// ---------------------------------------------------------------------------
// out[b,t,o] = leaky( hs[t,b,:] . W_out[o,:] + b_out[o] ), rows m = b*T + t.
// Grid (128,16), 256 threads = 8 waves; wave: 2 M-tiles x 4 N-tiles.
// ---------------------------------------------------------------------------
__global__ __launch_bounds__(256)
void out_proj_kernel(const unsigned short* __restrict__ hs,   // [T,B,H] bf16
                     const unsigned short* __restrict__ Wo,   // [O,H] bf16
                     const float* __restrict__ bout,
                     float* __restrict__ out) {               // [B,T,O] fp32
    const int w    = threadIdx.x >> 5;
    const int lane = threadIdx.x & 31;
    const int half = lane >> 4, lrow = lane & 15;
    const int m_base = blockIdx.x * 256 + w * 32;
    const int n_base = blockIdx.y * 64;

    floatx8 acc[4][2];
#pragma unroll
    for (int nt = 0; nt < 4; ++nt) { acc[nt][0] = floatx8{}; acc[nt][1] = floatx8{}; }

    const int krow = 16 * half;
    const int m0 = m_base + lrow, m1 = m0 + 16;               // m = b*T + t
    const unsigned short* a0p = hs + ((size_t)(m0 & (T_ - 1)) * B_ + (m0 >> 8)) * H_ + krow;
    const unsigned short* a1p = hs + ((size_t)(m1 & (T_ - 1)) * B_ + (m1 >> 8)) * H_ + krow;
    const unsigned short* bp[4];
#pragma unroll
    for (int nt = 0; nt < 4; ++nt) bp[nt] = Wo + (size_t)(n_base + 16 * nt + lrow) * H_ + krow;

    for (int k0 = 0; k0 < H_; k0 += 32) {
        bf16x16 a0 = ldbf(a0p + k0);
        bf16x16 a1 = ldbf(a1p + k0);
#pragma unroll
        for (int nt = 0; nt < 4; ++nt) {
            bf16x16 bm = ldbf(bp[nt] + k0);
            acc[nt][0] = __builtin_amdgcn_wmma_f32_16x16x32_bf16(false, a0, false, bm, (short)0, acc[nt][0], false, false);
            acc[nt][1] = __builtin_amdgcn_wmma_f32_16x16x32_bf16(false, a1, false, bm, (short)0, acc[nt][1], false, false);
        }
    }

#pragma unroll
    for (int nt = 0; nt < 4; ++nt) {
        int n = n_base + 16 * nt + lrow;
        float bb = bout[n];
#pragma unroll
        for (int mt = 0; mt < 2; ++mt) {
#pragma unroll
            for (int r = 0; r < 8; ++r) {
                int m = m_base + 16 * mt + 8 * half + r;
                float v = acc[nt][mt][r] + bb;
                out[(size_t)m * O_ + n] = (v >= 0.0f) ? v : 0.2f * v;
            }
        }
    }
}

// ---------------------------------------------------------------------------
extern "C" void kernel_launch(void* const* d_in, const int* in_sizes, int n_in,
                              void* d_out, int out_size, void* d_ws, size_t ws_size,
                              hipStream_t stream) {
    const float* z    = (const float*)d_in[0];
    // d_in[1] = sequence_length (==T_, compile-time constant here)
    const float* Wfh  = (const float*)d_in[2];
    const float* bfh  = (const float*)d_in[3];
    const float* Wfc  = (const float*)d_in[4];
    const float* bfc  = (const float*)d_in[5];
    const float* Wih  = (const float*)d_in[6];
    const float* bih  = (const float*)d_in[7];
    const float* Whh  = (const float*)d_in[8];
    const float* bhh  = (const float*)d_in[9];
    const float* x0   = (const float*)d_in[10];
    const float* Wout = (const float*)d_in[11];
    const float* bout = (const float*)d_in[12];
    float* out = (float*)d_out;

    // Workspace carve-up (all 256B aligned)
    char* ws = (char*)d_ws;
    size_t off = 0;
    auto carve = [&](size_t bytes) { char* p = ws + off; off = (off + bytes + 255) & ~(size_t)255; return p; };
    unsigned short* hs_bf = (unsigned short*)carve((size_t)T_ * B_ * H_ * 2);   // 64 MB
    unsigned short* Wc_bf = (unsigned short*)carve((size_t)G4H * H_ * 2);       //  8 MB
    unsigned short* Wh_bf = (unsigned short*)carve((size_t)G4H * H_ * 2);       //  8 MB
    unsigned short* Wo_bf = (unsigned short*)carve((size_t)O_ * H_ * 2);        //  2 MB
    float*          c_buf = (float*)carve((size_t)B_ * H_ * 4);
    unsigned short* hA    = (unsigned short*)carve((size_t)B_ * H_ * 2);        // holds h0, then ping-pong
    unsigned short* hB    = (unsigned short*)carve((size_t)B_ * H_ * 2);
    float*          u_b   = (float*)carve((size_t)G4H * 4);
    float*          bc_b  = (float*)carve((size_t)G4H * 4);

    // Phase 0: precompute (initial state, folded step-0 bias, bf16 weights)
    init_state_kernel<<<(B_ * H_ + 255) / 256, 256, 0, stream>>>(z, Wfh, bfh, Wfc, bfc, hA, c_buf);
    init_gatebias_kernel<<<(G4H + 255) / 256, 256, 0, stream>>>(x0, Wih, bih, bhh, u_b, bc_b);
    conv_rec_weights_kernel<<<(G4H * H_ + 255) / 256, 256, 0, stream>>>(Wih, Whh, Wc_bf, Wh_bf);
    conv_out_weights_kernel<<<(O_ * H_ + 255) / 256, 256, 0, stream>>>(Wout, Wo_bf);

    // Phase 1: sequential recurrence, one launch per timestep.
    // t==0: gates = h0 @ W_hh.T + u      (u folds x0@W_ih.T + both biases)
    // t>=1: gates = h  @ (W_ih+W_hh).T + (b_ih+b_hh)   (since x == h)
    for (int t = 0; t < T_; ++t) {
        const unsigned short* Ain = (t & 1) ? hB : hA;
        unsigned short*       Aout = (t & 1) ? hA : hB;
        const unsigned short* Wsel = (t == 0) ? Wh_bf : Wc_bf;
        const float*          bsel = (t == 0) ? u_b : bc_b;
        lstm_step_kernel<<<dim3(H_ / 64), dim3(512), 0, stream>>>(Ain, Wsel, bsel, c_buf, hs_bf, Aout, t);
    }

    // Phase 2: output projection + LeakyReLU(0.2)
    out_proj_kernel<<<dim3((B_ * T_) / 256, O_ / 64), dim3(256), 0, stream>>>(hs_bf, Wo_bf, bout, out);
}
